// DynamicRouting_56324201120340
// MI455X (gfx1250) — matrix operand order repaired and checked by reading, not compiled, not run on hardware
//
#include <hip/hip_runtime.h>
#include <hip/hip_bf16.h>

// ---------------- types ----------------
typedef __attribute__((ext_vector_type(16))) __bf16 v16bf;
typedef __attribute__((ext_vector_type(8)))  __bf16 v8bf;
typedef __attribute__((ext_vector_type(8)))  float  v8f;
typedef __attribute__((address_space(3))) unsigned short lds_ushort;

#define B_   64
#define N_   512
#define DIN  1024
#define DOUT 1024
#define M_   (B_ * N_)          // 32768 flattened GEMM rows
#define EPS_ 1e-8f

// ---------------- helpers ----------------
__device__ __forceinline__ unsigned short f32_to_bf16(float f) {
    unsigned int u = __float_as_uint(f);
    u += 0x7FFFu + ((u >> 16) & 1u);   // round-to-nearest-even
    return (unsigned short)(u >> 16);
}
__device__ __forceinline__ unsigned int pack_bf16x2(float a, float b) {
    return (unsigned int)f32_to_bf16(a) | ((unsigned int)f32_to_bf16(b) << 16);
}

// CDNA5 async global->LDS copy (ASYNCcnt-tracked), 16 bytes per lane.
__device__ __forceinline__ void async_copy16(const unsigned short* g, unsigned short* l) {
#if defined(__gfx1250__)
    lds_ushort* lp = (lds_ushort*)l;
    unsigned lds_off = (unsigned)(unsigned long)lp;
    asm volatile("global_load_async_to_lds_b128 %0, %1, off"
                 :: "v"(lds_off), "v"(g) : "memory");
#else
    *(uint4*)l = *(const uint4*)g;
#endif
}
template <int N>
__device__ __forceinline__ void async_wait() {
#if defined(__gfx1250__)
    asm volatile("s_wait_asynccnt %0" :: "i"(N) : "memory");
#endif
}

__device__ __forceinline__ float blockRedSum(float v, float* sred) {
    int tid = threadIdx.x;
    #pragma unroll
    for (int off = 16; off; off >>= 1) v += __shfl_xor(v, off, 32);
    if ((tid & 31) == 0) sred[tid >> 5] = v;
    __syncthreads();
    if (tid < 32) {
        int nw = blockDim.x >> 5;
        float x = (tid < nw) ? sred[tid] : 0.0f;
        #pragma unroll
        for (int off = 8; off; off >>= 1) x += __shfl_xor(x, off, 32);
        if (tid == 0) sred[0] = x;
    }
    __syncthreads();
    float r = sred[0];
    __syncthreads();
    return r;
}

__device__ __forceinline__ float blockRedMax(float v, float* sred) {
    int tid = threadIdx.x;
    #pragma unroll
    for (int off = 16; off; off >>= 1) v = fmaxf(v, __shfl_xor(v, off, 32));
    if ((tid & 31) == 0) sred[tid >> 5] = v;
    __syncthreads();
    if (tid < 32) {
        int nw = blockDim.x >> 5;
        float x = (tid < nw) ? sred[tid] : -__builtin_inff();
        #pragma unroll
        for (int off = 8; off; off >>= 1) x = fmaxf(x, __shfl_xor(x, off, 32));
        if (tid == 0) sred[0] = x;
    }
    __syncthreads();
    float r = sred[0];
    __syncthreads();
    return r;
}

// ---------------- 1) convert x (fp32) -> bf16, 8 elems/thread ----------------
__global__ void cvt_x_kernel(const float* __restrict__ x, uint4* __restrict__ xb, size_t n8) {
    size_t i = (size_t)blockIdx.x * blockDim.x + threadIdx.x;
    if (i >= n8) return;
    const float4* pv = (const float4*)x;
    float4 f0 = pv[i * 2 + 0];
    float4 f1 = pv[i * 2 + 1];
    uint4 o;
    o.x = pack_bf16x2(f0.x, f0.y);
    o.y = pack_bf16x2(f0.z, f0.w);
    o.z = pack_bf16x2(f1.x, f1.y);
    o.w = pack_bf16x2(f1.z, f1.w);
    xb[i] = o;
}

// ---------------- 2) transpose + convert W[k][n] -> Wt[n][k] bf16 ----------------
__global__ void cvt_wt_kernel(const float* __restrict__ W, unsigned short* __restrict__ Wt) {
    __shared__ unsigned short tile[32][33];
    int bx = blockIdx.x * 32;   // n base
    int by = blockIdx.y * 32;   // k base
    int tx = threadIdx.x;       // 0..31
    int ty = threadIdx.y;       // 0..7
    #pragma unroll
    for (int i = 0; i < 32; i += 8) {
        tile[ty + i][tx] = f32_to_bf16(W[(size_t)(by + ty + i) * DOUT + (bx + tx)]);
    }
    __syncthreads();
    #pragma unroll
    for (int i = 0; i < 32; i += 8) {
        Wt[(size_t)(bx + ty + i) * DIN + (by + tx)] = tile[tx][ty + i];
    }
}

// ---------------- 3) bf16 WMMA GEMM, double-buffered async-LDS B staging ----------------
// block = 256 threads (8 waves) -> tile 128(M) x 128(N).
// wave (mw = w&3, nw = w>>2) computes 32(M) x 64(N): 2 A-frags x 4 B-frags.
// K-step 64: 16 WMMA per step; B tile (128 x 64 bf16, 16 KB) double-buffered in LDS,
// filled by global_load_async_to_lds_b128 one step ahead (ASYNCcnt pipelining).
#define BK 64
#define KSTEPS (DIN / BK)     // 16
#define BSTRIDE 72            // ushorts/row: 144B -> 16B-aligned rows, conflict-free b128 reads

__global__ __launch_bounds__(256)
void gemm_bf16_kernel(const unsigned short* __restrict__ Xb,   // [M_][DIN] bf16 bits
                      const unsigned short* __restrict__ Wt,   // [DOUT][DIN] bf16 bits
                      float* __restrict__ xw)                  // [M_][DOUT] f32
{
    __shared__ unsigned short Bs[2][128][BSTRIDE];   // 36 KB

    const int K = DIN;
    int tid  = threadIdx.x;
    int wave = tid >> 5;
    int lane = tid & 31;
    int half = lane >> 4;      // 0 or 1
    int l15  = lane & 15;
    int mw   = wave & 3;       // 4 M-wave groups of 32 rows
    int nw   = wave >> 2;      // 2 N-wave groups of 64 cols

    int mBlock = blockIdx.x * 128;   // gridDim.x = M_/128  = 256
    int nBlock = blockIdx.y * 128;   // gridDim.y = DOUT/128 = 8
    int m0     = mBlock + mw * 32;
    int nWave  = nw * 64;            // within-block N offset

    v8f acc[2][4] = {};

    // A: lane holds row; per 32-K sub-step, chunks K = kk + half*8 and kk + half*8 + 16
    const unsigned short* arow0 = Xb + (size_t)(m0 +      l15) * K;
    const unsigned short* arow1 = Xb + (size_t)(m0 + 16 + l15) * K;

    // B staging: thread t owns row = t>>1, 64B half-row at skk = (t&1)*32 ushorts
    int srow = tid >> 1;
    int skk  = (tid & 1) * 32;
    const unsigned short* bsrc = Wt + (size_t)(nBlock + srow) * K + skk;

    auto issue = [&](int buf, int k) {
        #pragma unroll
        for (int c = 0; c < 4; ++c)
            async_copy16(bsrc + k + c * 8, &Bs[buf][srow][skk + c * 8]);
    };

    issue(0, 0);                                   // prologue: fill buffer 0

    for (int ks = 0; ks < KSTEPS; ++ks) {
        int k   = ks * BK;
        int cur = ks & 1;
        if (ks + 1 < KSTEPS) {
            issue(cur ^ 1, k + BK);                // DMA next tile under this step's WMMAs
            __builtin_prefetch(arow0 + k + BK, 0, 3);
            __builtin_prefetch(arow1 + k + BK, 0, 3);
            async_wait<4>();                       // oldest 4 (current buffer) retired
        } else {
            async_wait<0>();
        }
        __syncthreads();                           // Bs[cur] visible to all waves

        #pragma unroll
        for (int sub = 0; sub < 2; ++sub) {
            int kk = sub * 32;
            union { v16bf v; v8bf h[2]; } ua0, ua1;
            ua0.h[0] = *(const v8bf*)(arow0 + k + kk + half * 8);
            ua0.h[1] = *(const v8bf*)(arow0 + k + kk + half * 8 + 16);
            ua1.h[0] = *(const v8bf*)(arow1 + k + kk + half * 8);
            ua1.h[1] = *(const v8bf*)(arow1 + k + kk + half * 8 + 16);

            #pragma unroll
            for (int j = 0; j < 4; ++j) {
                const unsigned short* bp = &Bs[cur][nWave + j * 16 + l15][kk + half * 8];
                union { v16bf v; v8bf h[2]; } ub;
                ub.h[0] = *(const v8bf*)(bp);
                ub.h[1] = *(const v8bf*)(bp + 16);
                acc[0][j] = __builtin_amdgcn_wmma_f32_16x16x32_bf16(
                    false, ua0.v, false, ub.v, (short)0, acc[0][j], false, false);
                acc[1][j] = __builtin_amdgcn_wmma_f32_16x16x32_bf16(
                    false, ua1.v, false, ub.v, (short)0, acc[1][j], false, false);
            }
        }
        __syncthreads();   // all waves done reading Bs[cur] before it is re-filled
    }

    // C/D layout: lane -> col l15; VGPR r -> row r + half*8
    #pragma unroll
    for (int i = 0; i < 2; ++i) {
        #pragma unroll
        for (int j = 0; j < 4; ++j) {
            int col = nBlock + nWave + j * 16 + l15;
            #pragma unroll
            for (int r = 0; r < 8; ++r) {
                int m = m0 + i * 16 + r + half * 8;
                xw[(size_t)m * DOUT + col] = acc[i][j][r];
            }
        }
    }
}

// ---------------- 4) per-row sum/norm + initial r_sim ----------------
__global__ __launch_bounds__(256)
void rowstats_kernel(const float* __restrict__ xw,
                     float* __restrict__ rowsum, float* __restrict__ rownorm,
                     float* __restrict__ r_sim) {
    __shared__ float sred[32];
    int row = blockIdx.x;
    int tid = threadIdx.x;
    const float* xr = xw + (size_t)row * DOUT;
    float s = 0.0f, q = 0.0f;
    #pragma unroll
    for (int i = 0; i < 4; ++i) {
        float v = xr[tid + (i << 8)];
        s += v;
        q = fmaf(v, v, q);
    }
    s = blockRedSum(s, sred);
    q = blockRedSum(q, sred);
    if (tid == 0) {
        rowsum[row]  = s;
        rownorm[row] = fmaxf(sqrtf(q), EPS_);
        r_sim[row]   = (s == 0.0f) ? -__builtin_inff() : 0.0f;
    }
}

// ---------------- 5a) softmax over N_ per batch ----------------
__global__ __launch_bounds__(512)
void softmax_kernel(const float* __restrict__ r_sim, float* __restrict__ wgt) {
    __shared__ float sred[32];
    int b = blockIdx.x, tid = threadIdx.x;
    float r = r_sim[b * N_ + tid];
    float m = blockRedMax(r, sred);
    float e = expf(r - m);                  // -inf rows -> 0
    float den = blockRedSum(e, sred);
    wgt[b * N_ + tid] = e / den;
}

// ---------------- 5b) partial x_g over 64-row chunks ----------------
__global__ __launch_bounds__(256)
void xg_partial_kernel(const float* __restrict__ xw, const float* __restrict__ wgt,
                       float* __restrict__ xgp) {
    int chunk = blockIdx.x;                 // 0..7
    int b     = blockIdx.y;                 // 0..63
    int tid   = threadIdx.x;                // 0..255
    __shared__ float sw[64];
    const float* xwb = xw + ((size_t)b * N_ + chunk * 64) * DOUT;
    if (tid < 64) sw[tid] = wgt[b * N_ + chunk * 64 + tid];
    __syncthreads();
    float a0 = 0.f, a1 = 0.f, a2 = 0.f, a3 = 0.f;
    for (int n = 0; n < 64; ++n) {
        float w = sw[n];
        const float* xr = xwb + (size_t)n * DOUT;
        a0 = fmaf(w, xr[tid      ], a0);
        a1 = fmaf(w, xr[tid + 256], a1);
        a2 = fmaf(w, xr[tid + 512], a2);
        a3 = fmaf(w, xr[tid + 768], a3);
    }
    float* o = xgp + ((size_t)b * 8 + chunk) * DOUT;
    o[tid] = a0; o[tid + 256] = a1; o[tid + 512] = a2; o[tid + 768] = a3;
}

// ---------------- 5c) deterministic reduce of partials + g_norm ----------------
__global__ __launch_bounds__(256)
void xg_reduce_kernel(const float* __restrict__ xgp, float* __restrict__ xg,
                      float* __restrict__ gnorm, float* __restrict__ out, int write_out) {
    __shared__ float sred[32];
    int b = blockIdx.x, tid = threadIdx.x;
    const float* p = xgp + (size_t)b * 8 * DOUT;
    float vals[4];
    float q = 0.f;
    #pragma unroll
    for (int i = 0; i < 4; ++i) {
        int d = tid + (i << 8);
        float s = 0.f;
        #pragma unroll
        for (int c = 0; c < 8; ++c) s += p[(size_t)c * DOUT + d];
        vals[i] = s;
        q = fmaf(s, s, q);
        xg[(size_t)b * DOUT + d] = s;
    }
    q = blockRedSum(q, sred);
    if (tid == 0) gnorm[b] = fmaxf(sqrtf(q), EPS_);
    if (write_out) {
        #pragma unroll
        for (int i = 0; i < 4; ++i) out[(size_t)b * DOUT + tid + (i << 8)] = vals[i];
    }
}

// ---------------- 5d) num / r_sim update (one wave per row, 4 rows/wave) ----------------
__global__ __launch_bounds__(256)
void num_kernel(const float* __restrict__ xw, const float* __restrict__ xg,
                const float* __restrict__ gnorm, const float* __restrict__ rowsum,
                const float* __restrict__ rownorm, float* __restrict__ r_sim) {
    __shared__ float sxg[DOUT];
    int rc = blockIdx.x;                    // 0..15 (row chunk of 32)
    int b  = blockIdx.y;                    // 0..63
    int tid = threadIdx.x;
    #pragma unroll
    for (int i = 0; i < 4; ++i)
        sxg[tid + (i << 8)] = xg[(size_t)b * DOUT + tid + (i << 8)];
    __syncthreads();
    float gn = gnorm[b];
    int wv = tid >> 5, lane = tid & 31;
    #pragma unroll
    for (int rr = 0; rr < 4; ++rr) {
        int row = rc * 32 + wv * 4 + rr;    // 0..511
        const float* xr = xw + ((size_t)b * N_ + row) * DOUT;
        float s = 0.f;
        #pragma unroll 8
        for (int i = 0; i < 32; ++i) {
            int d = lane + (i << 5);
            s = fmaf(xr[d], sxg[d], s);
        }
        #pragma unroll
        for (int off = 16; off; off >>= 1) s += __shfl_xor(s, off, 32);
        if (lane == 0) {
            int g = b * N_ + row;
            r_sim[g] = (rowsum[g] == 0.0f) ? -__builtin_inff()
                                           : s / (rownorm[g] * gn);
        }
    }
}

// ---------------- launch ----------------
extern "C" void kernel_launch(void* const* d_in, const int* in_sizes, int n_in,
                              void* d_out, int out_size, void* d_ws, size_t ws_size,
                              hipStream_t stream) {
    const float* x = (const float*)d_in[0];   // [64,512,1024] f32
    const float* W = (const float*)d_in[1];   // [1,1024,1024] f32
    float* out = (float*)d_out;               // [64,1024] f32

    char* ws = (char*)d_ws;
    size_t off = 0;
    unsigned short* xb = (unsigned short*)(ws + off); off += (size_t)M_ * DIN * 2;
    unsigned short* wt = (unsigned short*)(ws + off); off += (size_t)DIN * DOUT * 2;
    float* xw      = (float*)(ws + off); off += (size_t)M_ * DOUT * 4;
    float* rowsum  = (float*)(ws + off); off += (size_t)M_ * 4;
    float* rownorm = (float*)(ws + off); off += (size_t)M_ * 4;
    float* r_sim   = (float*)(ws + off); off += (size_t)M_ * 4;
    float* wgt     = (float*)(ws + off); off += (size_t)M_ * 4;
    float* xgp     = (float*)(ws + off); off += (size_t)B_ * 8 * DOUT * 4;
    float* xg      = (float*)(ws + off); off += (size_t)B_ * DOUT * 4;
    float* gnorm   = (float*)(ws + off); off += (size_t)B_ * 4;
    (void)in_sizes; (void)n_in; (void)out_size; (void)ws_size;

    // 1) x -> bf16
    size_t n8 = (size_t)M_ * DIN / 8;
    cvt_x_kernel<<<(unsigned)((n8 + 255) / 256), 256, 0, stream>>>(x, (uint4*)xb, n8);

    // 2) W -> Wt (bf16, transposed)
    cvt_wt_kernel<<<dim3(DOUT / 32, DIN / 32), dim3(32, 8), 0, stream>>>(W, wt);

    // 3) GEMM via v_wmma_f32_16x16x32_bf16 + double-buffered async-LDS B staging
    gemm_bf16_kernel<<<dim3(M_ / 128, DOUT / 128), 256, 0, stream>>>(xb, wt, xw);

    // 4) row sums / norms / initial r_sim
    rowstats_kernel<<<M_, 256, 0, stream>>>(xw, rowsum, rownorm, r_sim);

    // 5) three routing iterations, grid-wide
    for (int it = 0; it < 3; ++it) {
        softmax_kernel<<<B_, 512, 0, stream>>>(r_sim, wgt);
        xg_partial_kernel<<<dim3(8, B_), 256, 0, stream>>>(xw, wgt, xgp);
        xg_reduce_kernel<<<B_, 256, 0, stream>>>(xgp, xg, gnorm, out, it == 2 ? 1 : 0);
        if (it < 2)
            num_kernel<<<dim3(16, B_), 256, 0, stream>>>(xw, xg, gnorm, rowsum, rownorm, r_sim);
    }
}
